// GhostLM_41197326303388
// MI455X (gfx1250) — compile-verified, matrix operand order checked by reading, not run on hardware
//
#include <hip/hip_runtime.h>

// GPT-style transformer forward (4 layers, D=1024, H=16, DFF=4096, B=2, T=1024,
// V=32000) for gfx1250. All GEMMs use V_WMMA_F32_16X16X32_BF16 (f32 accumulate).
// Dense GEMMs are LDS-staged with double buffering; the staging copies use
// GLOBAL_LOAD_ASYNC_TO_LDS_B128 (ASYNCcnt-tracked, s_wait_asynccnt).

#define LNUM   4
#define VSZ    32000
#define TSEQ   1024
#define DMODEL 1024
#define NHEAD  16
#define HDIM   64
#define DFFN   4096
#define BSZ    2
#define MROWS  (BSZ * TSEQ)   // 2048

// LDS-staged GEMM tiling
#define BM   128
#define BN   128
#define BK   32
#define PADS 40               // LDS row stride in bf16 elems (80B: 16B-aligned, conflict-free)

typedef __attribute__((ext_vector_type(16))) __bf16 v16bf;
typedef __attribute__((ext_vector_type(8)))  float  v8f;
typedef __attribute__((ext_vector_type(4)))  int    v4i;

#if __has_builtin(__builtin_amdgcn_global_load_async_to_lds_b128)
#define HAVE_ASYNC_LDS 1
typedef __attribute__((address_space(1))) v4i gv4i;   // global-side b128
typedef __attribute__((address_space(3))) v4i lv4i;   // LDS-side b128
#else
#define HAVE_ASYNC_LDS 0
#endif

// ---------------------------------------------------------------- helpers ---

__device__ __forceinline__ __bf16 to_bf16(float f) {
  // round-to-nearest-even fp32 -> bf16
  unsigned u = __builtin_bit_cast(unsigned, f);
  u += 0x7FFFu + ((u >> 16) & 1u);
  unsigned short h = (unsigned short)(u >> 16);
  return __builtin_bit_cast(__bf16, h);
}

// A-fragment (16x32 bf16, MxK) per ISA 7.12.2:
//   lanes 0-15 -> M = lane, K groups {0..7,16..23}; lanes 16-31 -> K {8..15,24..31}
//   VGPR v holds packed pair (k, k+1). The same striping serves the B operand
//   when B is given as W (NxK row-major, i.e. B = W^T): lane selects N.
//   All reads are K-contiguous -> vectorizes to b128 (global or ds).
__device__ __forceinline__ v16bf load_frag_k(const __bf16* __restrict__ p, int ld) {
  int lane = threadIdx.x & 31;
  int g    = lane >> 4;                       // half-wave selects K sub-block
  const __bf16* row = p + (size_t)(lane & 15) * ld;
  v16bf f;
#pragma unroll
  for (int v = 0; v < 8; ++v) {
    int k = ((v & 3) << 1) + ((v >> 2) << 4) + (g << 3);
    f[2 * v]     = row[k];
    f[2 * v + 1] = row[k + 1];
  }
  return f;
}

// Stage a BMxBK (=128x32) bf16 tile into LDS: 512 16-byte segments, 256 threads
// x2. Async (ASYNCcnt) when available, else direct load+store (DScnt/barrier).
__device__ __forceinline__ void stage_tile(const __bf16* __restrict__ src, int ld,
                                           __bf16* dst) {
  int t = threadIdx.x;
#pragma unroll
  for (int i = 0; i < 2; ++i) {
    int s   = t + i * 256;
    int row = s >> 2;
    int cs  = (s & 3) * 8;                    // bf16 elems (16B granules)
    const __bf16* g = src + (size_t)row * ld + cs;
    __bf16* l = dst + row * PADS + cs;
#if HAVE_ASYNC_LDS
    __builtin_amdgcn_global_load_async_to_lds_b128(
        (gv4i*)(uintptr_t)g, (lv4i*)(unsigned)(uintptr_t)l, 0, 0);
#else
    *(v4i*)l = *(const v4i*)g;
#endif
  }
}

template <int N>
__device__ __forceinline__ void wait_async() {
#if HAVE_ASYNC_LDS
#if __has_builtin(__builtin_amdgcn_s_wait_asynccnt)
  __builtin_amdgcn_s_wait_asynccnt(N);
#else
  if (N == 0) asm volatile("s_wait_asynccnt 0x0" ::: "memory");
  else        asm volatile("s_wait_asynccnt 0x4" ::: "memory");
#endif
#endif
}

// ----------------------------------------------------------------- kernels --

__global__ void embed_kernel(const int* __restrict__ idx,
                             const float* __restrict__ wte,
                             const float* __restrict__ wpe,
                             float* __restrict__ x) {
  int r = blockIdx.x;                // b*T + t
  int t = r % TSEQ;
  int tok = idx[r];
  const float* te = wte + (size_t)tok * DMODEL;
  const float* pe = wpe + (size_t)t   * DMODEL;
  float* xr = x + (size_t)r * DMODEL;
  for (int i = threadIdx.x; i < DMODEL; i += blockDim.x) xr[i] = te[i] + pe[i];
}

__global__ void ln_bf16_kernel(const float* __restrict__ x,
                               const float* __restrict__ w,
                               const float* __restrict__ b,
                               __bf16* __restrict__ out) {
  int r = blockIdx.x;
  const float* row = x + (size_t)r * DMODEL;
  __bf16* orow = out + (size_t)r * DMODEL;
  int tid = threadIdx.x;
  __shared__ float s1[256], s2[256];
  float a = 0.f, a2 = 0.f;
  for (int i = tid; i < DMODEL; i += 256) { float v = row[i]; a += v; a2 += v * v; }
  s1[tid] = a; s2[tid] = a2; __syncthreads();
  for (int s = 128; s > 0; s >>= 1) {
    if (tid < s) { s1[tid] += s1[tid + s]; s2[tid] += s2[tid + s]; }
    __syncthreads();
  }
  float mu   = s1[0] * (1.0f / DMODEL);
  float var  = s2[0] * (1.0f / DMODEL) - mu * mu;
  float rstd = rsqrtf(var + 1e-5f);
  for (int i = tid; i < DMODEL; i += 256)
    orow[i] = to_bf16((row[i] - mu) * rstd * w[i] + b[i]);
}

__global__ void f32_to_bf16_kernel(const float* __restrict__ in,
                                   __bf16* __restrict__ out, size_t n) {
  size_t stride = (size_t)gridDim.x * blockDim.x;
  for (size_t i = (size_t)blockIdx.x * blockDim.x + threadIdx.x; i < n; i += stride)
    out[i] = to_bf16(in[i]);
}

// qkv (bf16, rows (b,t) x 3D) -> Q,K: (B,H,T,HD); V: (B,H,HD,T) transposed so
// the P@V B-operand is K-contiguous (W-form fragment loads, b128-vectorizable).
__global__ void split_qkv_kernel(const __bf16* __restrict__ qkv,
                                 __bf16* __restrict__ q,
                                 __bf16* __restrict__ k,
                                 __bf16* __restrict__ vt) {
  size_t n = (size_t)MROWS * DMODEL;
  size_t stride = (size_t)gridDim.x * blockDim.x;
  for (size_t e = (size_t)blockIdx.x * blockDim.x + threadIdx.x; e < n; e += stride) {
    int d = (int)(e % HDIM);
    int h = (int)((e / HDIM) % NHEAD);
    size_t row = e / DMODEL;                       // b*T + t
    int t = (int)(row % TSEQ);
    int b = (int)(row / TSEQ);
    size_t bh  = (size_t)b * NHEAD + h;
    size_t src = row * (size_t)(3 * DMODEL) + (size_t)h * HDIM + d;
    size_t dst = (bh * TSEQ + t) * HDIM + d;       // (B,H,T,HD)
    q[dst] = qkv[src];
    k[dst] = qkv[src + DMODEL];
    vt[(bh * HDIM + d) * TSEQ + t] = qkv[src + 2 * DMODEL];   // (B,H,HD,T)
  }
}

__global__ void merge_heads_kernel(const float* __restrict__ y,
                                   __bf16* __restrict__ out) {
  // y: (B,H,T,HD) f32 -> out: (B*T, D) bf16
  size_t n = (size_t)MROWS * DMODEL;
  size_t stride = (size_t)gridDim.x * blockDim.x;
  for (size_t e = (size_t)blockIdx.x * blockDim.x + threadIdx.x; e < n; e += stride) {
    int d = (int)(e % HDIM);
    int h = (int)((e / HDIM) % NHEAD);
    size_t row = e / DMODEL;
    int t = (int)(row % TSEQ);
    int b = (int)(row / TSEQ);
    out[e] = to_bf16(y[(((size_t)b * NHEAD + h) * TSEQ + t) * HDIM + d]);
  }
}

__global__ void softmax_causal_kernel(const float* __restrict__ S,
                                      __bf16* __restrict__ P) {
  int r = blockIdx.x;                 // bh*T + q
  int q = r & (TSEQ - 1);
  const float* srow = S + (size_t)r * TSEQ;
  __bf16* prow = P + (size_t)r * TSEQ;
  int tid = threadIdx.x;
  __shared__ float red[256];
  float m = -1e30f;
  for (int j = tid; j <= q; j += 256) m = fmaxf(m, srow[j]);
  red[tid] = m; __syncthreads();
  for (int s = 128; s > 0; s >>= 1) {
    if (tid < s) red[tid] = fmaxf(red[tid], red[tid + s]);
    __syncthreads();
  }
  m = red[0]; __syncthreads();
  float sum = 0.f;
  for (int j = tid; j <= q; j += 256) sum += __expf(srow[j] - m);
  red[tid] = sum; __syncthreads();
  for (int s = 128; s > 0; s >>= 1) {
    if (tid < s) red[tid] += red[tid + s];
    __syncthreads();
  }
  float inv = 1.0f / red[0];
  for (int j = tid; j < TSEQ; j += 256)
    prow[j] = to_bf16((j <= q) ? __expf(srow[j] - m) * inv : 0.f);
}

// LDS-staged double-buffered WMMA GEMM: C[M,N] = op(A[M,K] @ W^T + bias (+C)).
//   A: bf16 row-major (lda); W: bf16 NxK row-major (ldb).
//   OUTBF=0: C is f32 (optional residual accumulate). OUTBF=1: C is bf16.
//   256 threads = 8 waves (4x2), block tile 128x128, wave tile 32x64 (2x4
//   WMMA frags -> 8 v_wmma per K=32 step). grid = (N/128, M/128).
template <bool BIAS, bool RESID, bool GELU, bool OUTBF>
__global__ __launch_bounds__(256) void wmma_gemm_lds_kernel(
    const __bf16* __restrict__ A, const __bf16* __restrict__ W,
    const float* __restrict__ bias, void* __restrict__ Cv,
    int K, int lda, int ldb, int ldc) {
  __shared__ __bf16 sA[2][BM * PADS];
  __shared__ __bf16 sB[2][BN * PADS];
  int m0 = blockIdx.y * BM;
  int n0 = blockIdx.x * BN;
  int w  = threadIdx.x >> 5;
  int wm = (w >> 1) * 32;          // wave row origin within block tile
  int wn = (w & 1) * 64;           // wave col origin within block tile

  v8f acc[2][4];
#pragma unroll
  for (int i = 0; i < 2; ++i)
#pragma unroll
    for (int j = 0; j < 4; ++j) acc[i][j] = {};

  // prologue: stage K-tile 0
  stage_tile(A + (size_t)m0 * lda, lda, sA[0]);
  stage_tile(W + (size_t)n0 * ldb, ldb, sB[0]);

  int nsteps = K / BK;
  for (int s = 0; s < nsteps; ++s) {
    int cur = s & 1;
    if (s + 1 < nsteps) {
      // stage next K-tile into the alternate buffer (4 async issues/thread),
      // then wait only for the *current* tile's copies (<= 4 outstanding).
      stage_tile(A + (size_t)m0 * lda + (s + 1) * BK, lda, sA[cur ^ 1]);
      stage_tile(W + (size_t)n0 * ldb + (s + 1) * BK, ldb, sB[cur ^ 1]);
      wait_async<4>();
    } else {
      wait_async<0>();
    }
    __syncthreads();

    v16bf af[2], bfr[4];
#pragma unroll
    for (int i = 0; i < 2; ++i)
      af[i] = load_frag_k(sA[cur] + (wm + 16 * i) * PADS, PADS);
#pragma unroll
    for (int j = 0; j < 4; ++j)
      bfr[j] = load_frag_k(sB[cur] + (wn + 16 * j) * PADS, PADS);
#pragma unroll
    for (int i = 0; i < 2; ++i)
#pragma unroll
      for (int j = 0; j < 4; ++j)
        acc[i][j] = __builtin_amdgcn_wmma_f32_16x16x32_bf16(
            false, af[i], false, bfr[j], (short)0, acc[i][j], false, false);
    __syncthreads();                 // all waves done reading before overwrite
  }

  int lane = threadIdx.x & 31;
  int cn = lane & 15;
  int g  = lane >> 4;
  auto emit = [&](int r, int c, float val) {
    if (BIAS)  val += bias[c];
    if (GELU)  val = 0.5f * val * (1.0f + erff(val * 0.70710678118654752f));
    size_t o = (size_t)r * ldc + c;
    if (OUTBF) {
      ((__bf16*)Cv)[o] = to_bf16(val);
    } else {
      float* Cf = (float*)Cv;
      if (RESID) val += Cf[o];
      Cf[o] = val;
    }
  };
#pragma unroll
  for (int i = 0; i < 2; ++i)
#pragma unroll
    for (int j = 0; j < 4; ++j)
#pragma unroll
      for (int v = 0; v < 8; ++v)
        emit(m0 + wm + 16 * i + v + (g << 3), n0 + wn + 16 * j + cn, acc[i][j][v]);
}

// P@V per head: Y_h[T,HD] = P[T,T] @ V_h. V stored transposed (HD x T), so the
// B operand uses the W-form (K-contiguous) fragment loader.
//   128 threads = 4 waves, block tile 64x64; batched over b*h (grid.z).
__global__ __launch_bounds__(128) void wmma_pv_kernel(
    const __bf16* __restrict__ P, const __bf16* __restrict__ Vt,
    float* __restrict__ Y) {
  size_t bh = blockIdx.z;
  const __bf16* Ph = P  + bh * (size_t)TSEQ * TSEQ;
  const __bf16* Vh = Vt + bh * (size_t)HDIM * TSEQ;    // (HD, T) row-major
  float* Yh = Y + bh * (size_t)TSEQ * HDIM;            // (T, HD)
  int w  = threadIdx.x >> 5;
  int m0 = blockIdx.y * 64 + (w >> 1) * 32;            // query rows
  int n0 = (w & 1) * 32;                               // head-dim cols (N=64)
  v8f c00 = {}, c01 = {}, c10 = {}, c11 = {};
  for (int k0 = 0; k0 < TSEQ; k0 += 32) {
    v16bf a0 = load_frag_k(Ph + (size_t)m0 * TSEQ + k0, TSEQ);
    v16bf a1 = load_frag_k(Ph + (size_t)(m0 + 16) * TSEQ + k0, TSEQ);
    v16bf b0 = load_frag_k(Vh + (size_t)n0 * TSEQ + k0, TSEQ);
    v16bf b1 = load_frag_k(Vh + (size_t)(n0 + 16) * TSEQ + k0, TSEQ);
    c00 = __builtin_amdgcn_wmma_f32_16x16x32_bf16(false, a0, false, b0, (short)0, c00, false, false);
    c01 = __builtin_amdgcn_wmma_f32_16x16x32_bf16(false, a0, false, b1, (short)0, c01, false, false);
    c10 = __builtin_amdgcn_wmma_f32_16x16x32_bf16(false, a1, false, b0, (short)0, c10, false, false);
    c11 = __builtin_amdgcn_wmma_f32_16x16x32_bf16(false, a1, false, b1, (short)0, c11, false, false);
  }
  int lane = threadIdx.x & 31;
  int cn = lane & 15;
  int g  = lane >> 4;
#pragma unroll
  for (int v = 0; v < 8; ++v) {
    int mr = v + (g << 3);
    Yh[(size_t)(m0 + mr)      * HDIM + n0 + cn]      = c00[v];
    Yh[(size_t)(m0 + mr)      * HDIM + n0 + 16 + cn] = c01[v];
    Yh[(size_t)(m0 + 16 + mr) * HDIM + n0 + cn]      = c10[v];
    Yh[(size_t)(m0 + 16 + mr) * HDIM + n0 + 16 + cn] = c11[v];
  }
}

// Attention scores: S[bh] = Q_h @ K_h^T * scale; 64x64 blocks fully above the
// causal diagonal are skipped (softmax only reads j <= q).
__global__ __launch_bounds__(128) void attn_scores_kernel(
    const __bf16* __restrict__ Q, const __bf16* __restrict__ Kc,
    float* __restrict__ S, float scale) {
  if (blockIdx.x > blockIdx.y) return;   // whole tile masked
  size_t bh = blockIdx.z;
  const __bf16* Qh = Q  + bh * (size_t)TSEQ * HDIM;
  const __bf16* Kh = Kc + bh * (size_t)TSEQ * HDIM;
  float* Sh = S + bh * (size_t)TSEQ * TSEQ;
  int w  = threadIdx.x >> 5;
  int m0 = blockIdx.y * 64 + (w >> 1) * 32;
  int n0 = blockIdx.x * 64 + (w & 1) * 32;
  v8f c00 = {}, c01 = {}, c10 = {}, c11 = {};
#pragma unroll
  for (int k0 = 0; k0 < HDIM; k0 += 32) {
    v16bf a0 = load_frag_k(Qh + (size_t)m0 * HDIM + k0, HDIM);
    v16bf a1 = load_frag_k(Qh + (size_t)(m0 + 16) * HDIM + k0, HDIM);
    v16bf b0 = load_frag_k(Kh + (size_t)n0 * HDIM + k0, HDIM);
    v16bf b1 = load_frag_k(Kh + (size_t)(n0 + 16) * HDIM + k0, HDIM);
    c00 = __builtin_amdgcn_wmma_f32_16x16x32_bf16(false, a0, false, b0, (short)0, c00, false, false);
    c01 = __builtin_amdgcn_wmma_f32_16x16x32_bf16(false, a0, false, b1, (short)0, c01, false, false);
    c10 = __builtin_amdgcn_wmma_f32_16x16x32_bf16(false, a1, false, b0, (short)0, c10, false, false);
    c11 = __builtin_amdgcn_wmma_f32_16x16x32_bf16(false, a1, false, b1, (short)0, c11, false, false);
  }
  int lane = threadIdx.x & 31;
  int cn = lane & 15;
  int g  = lane >> 4;
#pragma unroll
  for (int v = 0; v < 8; ++v) {
    int mr = v + (g << 3);
    Sh[(size_t)(m0 + mr)      * TSEQ + n0 + cn]      = c00[v] * scale;
    Sh[(size_t)(m0 + mr)      * TSEQ + n0 + 16 + cn] = c01[v] * scale;
    Sh[(size_t)(m0 + 16 + mr) * TSEQ + n0 + cn]      = c10[v] * scale;
    Sh[(size_t)(m0 + 16 + mr) * TSEQ + n0 + 16 + cn] = c11[v] * scale;
  }
}

// ------------------------------------------------------------------- host ---

extern "C" void kernel_launch(void* const* d_in, const int* in_sizes, int n_in,
                              void* d_out, int out_size, void* d_ws, size_t ws_size,
                              hipStream_t stream) {
  (void)in_sizes; (void)n_in; (void)out_size; (void)ws_size;
  const int*   idx    = (const int*)  d_in[0];
  const float* wte    = (const float*)d_in[1];
  const float* wpe    = (const float*)d_in[2];
  const float* ln1_w  = (const float*)d_in[3];
  const float* ln1_b  = (const float*)d_in[4];
  const float* qkv_w  = (const float*)d_in[5];
  const float* qkv_b  = (const float*)d_in[6];
  const float* proj_w = (const float*)d_in[7];
  const float* proj_b = (const float*)d_in[8];
  const float* ln2_w  = (const float*)d_in[9];
  const float* ln2_b  = (const float*)d_in[10];
  const float* fc1_w  = (const float*)d_in[11];
  const float* fc1_b  = (const float*)d_in[12];
  const float* fc2_w  = (const float*)d_in[13];
  const float* fc2_b  = (const float*)d_in[14];
  const float* lnf_w  = (const float*)d_in[15];
  const float* lnf_b  = (const float*)d_in[16];
  float* out = (float*)d_out;

  char* wsp = (char*)d_ws;
  size_t off = 0;
  auto alloc = [&](size_t bytes) -> void* {
    void* p = wsp + off;
    off += (bytes + 255) & ~(size_t)255;
    return p;
  };
  float*  x    = (float*) alloc((size_t)MROWS * DMODEL * 4);       // residual
  __bf16* xa   = (__bf16*)alloc((size_t)MROWS * DMODEL * 2);       // bf16 A stage
  __bf16* ha   = (__bf16*)alloc((size_t)MROWS * DFFN * 2);         // bf16 FFN hidden
  __bf16* wb   = (__bf16*)alloc((size_t)VSZ * DMODEL * 2);         // bf16 weights
  __bf16* qkvb = (__bf16*)alloc((size_t)MROWS * 3 * DMODEL * 2);   // bf16 qkv
  __bf16* qb   = (__bf16*)alloc((size_t)MROWS * DMODEL * 2);       // (B,H,T,HD)
  __bf16* kb   = (__bf16*)alloc((size_t)MROWS * DMODEL * 2);       // (B,H,T,HD)
  __bf16* vtb  = (__bf16*)alloc((size_t)MROWS * DMODEL * 2);       // (B,H,HD,T)
  float*  S    = (float*) alloc((size_t)BSZ * NHEAD * TSEQ * TSEQ * 4);
  __bf16* P    = (__bf16*)alloc((size_t)BSZ * NHEAD * TSEQ * TSEQ * 2);
  float*  y    = (float*) alloc((size_t)MROWS * DMODEL * 4);       // (B,H,T,HD)

  const float scale = 0.125f;   // 1/sqrt(HD=64)
  auto blks = [](size_t n) { return (int)((n + 255) / 256); };

  embed_kernel<<<MROWS, 256, 0, stream>>>(idx, wte, wpe, x);

  for (int l = 0; l < LNUM; ++l) {
    // ---- attention ----
    ln_bf16_kernel<<<MROWS, 256, 0, stream>>>(x, ln1_w + l * DMODEL, ln1_b + l * DMODEL, xa);
    {
      size_t n = (size_t)3 * DMODEL * DMODEL;
      f32_to_bf16_kernel<<<blks(n), 256, 0, stream>>>(qkv_w + (size_t)l * n, wb, n);
    }
    wmma_gemm_lds_kernel<true, false, false, true>
        <<<dim3(3 * DMODEL / BN, MROWS / BM, 1), 256, 0, stream>>>(
            xa, wb, qkv_b + (size_t)l * 3 * DMODEL, qkvb,
            DMODEL, DMODEL, DMODEL, 3 * DMODEL);
    split_qkv_kernel<<<blks((size_t)MROWS * DMODEL), 256, 0, stream>>>(qkvb, qb, kb, vtb);
    attn_scores_kernel<<<dim3(TSEQ / 64, TSEQ / 64, BSZ * NHEAD), 128, 0, stream>>>(
        qb, kb, S, scale);
    softmax_causal_kernel<<<BSZ * NHEAD * TSEQ, 256, 0, stream>>>(S, P);
    wmma_pv_kernel<<<dim3(1, TSEQ / 64, BSZ * NHEAD), 128, 0, stream>>>(P, vtb, y);
    merge_heads_kernel<<<blks((size_t)MROWS * DMODEL), 256, 0, stream>>>(y, xa);
    {
      size_t n = (size_t)DMODEL * DMODEL;
      f32_to_bf16_kernel<<<blks(n), 256, 0, stream>>>(proj_w + (size_t)l * n, wb, n);
    }
    wmma_gemm_lds_kernel<true, true, false, false>
        <<<dim3(DMODEL / BN, MROWS / BM, 1), 256, 0, stream>>>(
            xa, wb, proj_b + (size_t)l * DMODEL, x, DMODEL, DMODEL, DMODEL, DMODEL);

    // ---- MLP ----
    ln_bf16_kernel<<<MROWS, 256, 0, stream>>>(x, ln2_w + l * DMODEL, ln2_b + l * DMODEL, xa);
    {
      size_t n = (size_t)DFFN * DMODEL;
      f32_to_bf16_kernel<<<blks(n), 256, 0, stream>>>(fc1_w + (size_t)l * n, wb, n);
    }
    wmma_gemm_lds_kernel<true, false, true, true>
        <<<dim3(DFFN / BN, MROWS / BM, 1), 256, 0, stream>>>(
            xa, wb, fc1_b + (size_t)l * DFFN, ha, DMODEL, DMODEL, DMODEL, DFFN);
    {
      size_t n = (size_t)DMODEL * DFFN;
      f32_to_bf16_kernel<<<blks(n), 256, 0, stream>>>(fc2_w + (size_t)l * n, wb, n);
    }
    wmma_gemm_lds_kernel<true, true, false, false>
        <<<dim3(DMODEL / BN, MROWS / BM, 1), 256, 0, stream>>>(
            ha, wb, fc2_b + (size_t)l * DMODEL, x, DFFN, DFFN, DFFN, DMODEL);
  }

  // ---- final LN + weight-tied logits ----
  ln_bf16_kernel<<<MROWS, 256, 0, stream>>>(x, lnf_w, lnf_b, xa);
  {
    size_t n = (size_t)VSZ * DMODEL;
    f32_to_bf16_kernel<<<blks(n), 256, 0, stream>>>(wte, wb, n);
  }
  wmma_gemm_lds_kernel<false, false, false, false>
      <<<dim3(VSZ / BN, MROWS / BM, 1), 256, 0, stream>>>(
          xa, wb, nullptr, out, DMODEL, DMODEL, DMODEL, VSZ);
}
